// CharsEncoder_88656714924243
// MI455X (gfx1250) — matrix operand order repaired and checked by reading, not compiled, not run on hardware
//
#include <hip/hip_runtime.h>
#include <hip/hip_bf16.h>

// ---------------------------------------------------------------------------
// CharsEncoder forward for MI455X (gfx1250, wave32, WMMA).
// All GEMMs (QKV/O, QK^T, attn@V, FFN, FC) go through one tiled WMMA kernel
// using v_wmma_f32_16x16x32_f16 with f32 accumulation; fp32 activations are
// converted to f16 while staging into LDS. The K-loop is software-pipelined:
// next tile's global loads are issued while the current tile's WMMAs run.
// Attention is processed per batch element so the 67MB logits/aw scratch
// stays resident in the 192MB L2.
// ---------------------------------------------------------------------------

typedef _Float16 half_t;
typedef __attribute__((ext_vector_type(16))) _Float16 v16h;
typedef __attribute__((ext_vector_type(4)))  _Float16 v4h;
typedef __attribute__((ext_vector_type(2)))  _Float16 v2h;
typedef __attribute__((ext_vector_type(8)))  float    v8f;
typedef __attribute__((ext_vector_type(4)))  unsigned int v4u;

#define D_   512
#define H_   8
#define HD_  64
#define DFF_ 2048
#define B_   4
#define S_   256
#define S4_  1024
#define HG_  2
#define EMB_ 512

// GEMM tiling: block = 256 threads = 8 waves; tile 128x128x32.
// Each wave: 32x64 C-tile = 2x4 WMMA 16x16 accumulators.
#define BM 128
#define BN 128
#define BK 32
// LDS row stride in halves: 40 halves = 80B = 20 banks.
//  - rows are 16B-aligned -> fragment loads are ds_load_b128
//  - 20*i mod 64 is distinct for i=0..15 -> conflict-free fragment reads
#define LDSS 40

union FragU { v16h v; v4u q[2]; };

// ---------------------------------------------------------------------------
// Generic batched GEMM: C = act(alpha * A @ B + bias)
//   A: M x K, row-major, leading dim lda, z-stride sAz
//   B: BNXK==false: K x N row-major (weights / V); true: N x K row-major (QK^T)
//   C: M x N, leading dim ldc, z-stride sCz
// Requirements (all call sites): M % 128 == 0, K % 32 == 0, N % 4 == 0.
// N may be partial vs BN (attn@V uses N=64) and is guarded per float4 / column.
// ---------------------------------------------------------------------------
template<int ACT, bool BNXK>
__global__ __launch_bounds__(256) void k_gemm(
    const float* __restrict__ A, int lda, long long sAz,
    const float* __restrict__ B, int ldb, long long sBz,
    float* __restrict__ C, int ldc, long long sCz,
    int M, int N, int K, const float* __restrict__ bias, float alpha)
{
  __shared__ half_t lA[BM * LDSS];
  __shared__ half_t lB[BN * LDSS];

  A += (long long)blockIdx.z * sAz;
  B += (long long)blockIdx.z * sBz;
  C += (long long)blockIdx.z * sCz;

  const int tid  = threadIdx.x;
  const int lane = tid & 31;
  const int wid  = tid >> 5;
  const int l15  = lane & 15;
  const int kh   = lane >> 4;          // 0: lanes 0-15, 1: lanes 16-31

  const int m0 = blockIdx.y * BM;
  const int n0 = blockIdx.x * BN;
  const int wm = (wid >> 1) * 32;      // wave row offset within tile
  const int wn = (wid & 1) * 64;       // wave col offset within tile

  v8f acc[2][4];
#pragma unroll
  for (int i = 0; i < 2; ++i)
#pragma unroll
    for (int j = 0; j < 4; ++j)
#pragma unroll
      for (int r = 0; r < 8; ++r) acc[i][j][r] = 0.0f;

  const unsigned int* lA32 = (const unsigned int*)lA;
  const unsigned int* lB32 = (const unsigned int*)lB;

  // ---- staging helpers (register tile <-> global / LDS) -------------------
  auto loadA = [&](int k0, float4 ra[4]) {
#pragma unroll
    for (int it = 0; it < 4; ++it) {
      int idx = tid + it * 256;          // 0..1023
      int r = idx >> 3;                  // row 0..127
      int c = (idx & 7) << 2;            // k 0,4,..,28
      ra[it] = *(const float4*)(A + (long long)(m0 + r) * lda + (k0 + c));
    }
  };
  auto loadB = [&](int k0, float4 rb[4]) {
    if (BNXK) {
#pragma unroll
      for (int it = 0; it < 4; ++it) {
        int idx = tid + it * 256;
        int r = idx >> 3;                // n_local 0..127
        int c = (idx & 7) << 2;
        int ng = n0 + r;
        rb[it] = make_float4(0.f, 0.f, 0.f, 0.f);
        if (ng < N) rb[it] = *(const float4*)(B + (long long)ng * ldb + (k0 + c));
      }
    } else {
      // two adjacent k-rows per thread (k, k+1) at the same columns
#pragma unroll
      for (int it = 0; it < 2; ++it) {
        int idx = tid + it * 256;        // 0..511
        int k = (idx >> 5) << 1;         // 0,2,..,30
        int c = (idx & 31) << 2;         // n_local 0..124
        rb[2 * it + 0] = make_float4(0.f, 0.f, 0.f, 0.f);
        rb[2 * it + 1] = make_float4(0.f, 0.f, 0.f, 0.f);
        if (n0 + c < N) {                // uniform per float4 (N % 4 == 0)
          rb[2 * it + 0] = *(const float4*)(B + (long long)(k0 + k + 0) * ldb + (n0 + c));
          rb[2 * it + 1] = *(const float4*)(B + (long long)(k0 + k + 1) * ldb + (n0 + c));
        }
      }
    }
  };
  auto storeA = [&](const float4 ra[4]) {
#pragma unroll
    for (int it = 0; it < 4; ++it) {
      int idx = tid + it * 256;
      int r = idx >> 3;
      int c = (idx & 7) << 2;
      v4h hv = { (half_t)ra[it].x, (half_t)ra[it].y, (half_t)ra[it].z, (half_t)ra[it].w };
      *(v4h*)&lA[r * LDSS + c] = hv;     // ds_store_b64
    }
  };
  auto storeB = [&](const float4 rb[4]) {
    if (BNXK) {
#pragma unroll
      for (int it = 0; it < 4; ++it) {
        int idx = tid + it * 256;
        int r = idx >> 3;
        int c = (idx & 7) << 2;
        v4h hv = { (half_t)rb[it].x, (half_t)rb[it].y, (half_t)rb[it].z, (half_t)rb[it].w };
        *(v4h*)&lB[r * LDSS + c] = hv;
      }
    } else {
      // transpose while storing: rows (k,k+1) x 4 cols -> four packed b32
      // stores; halves (k,k+1) are contiguous in the [n][k] LDS layout.
#pragma unroll
      for (int it = 0; it < 2; ++it) {
        int idx = tid + it * 256;
        int k = (idx >> 5) << 1;         // even
        int c = (idx & 31) << 2;
        const float4 lo = rb[2 * it + 0], hi = rb[2 * it + 1];
        v2h p0 = { (half_t)lo.x, (half_t)hi.x };
        v2h p1 = { (half_t)lo.y, (half_t)hi.y };
        v2h p2 = { (half_t)lo.z, (half_t)hi.z };
        v2h p3 = { (half_t)lo.w, (half_t)hi.w };
        *(v2h*)&lB[(c + 0) * LDSS + k] = p0;   // ds_store_b32 each
        *(v2h*)&lB[(c + 1) * LDSS + k] = p1;
        *(v2h*)&lB[(c + 2) * LDSS + k] = p2;
        *(v2h*)&lB[(c + 3) * LDSS + k] = p3;
      }
    }
  };

  // ---- software-pipelined K loop ------------------------------------------
  float4 ra[4], rb[4];
  loadA(0, ra);
  loadB(0, rb);

  for (int k0 = 0; k0 < K; k0 += BK) {
    storeA(ra);
    storeB(rb);
    __syncthreads();

    const bool more = (k0 + BK) < K;
    float4 na[4], nb[4];
    if (more) {            // issue next tile's global loads; they complete
      loadA(k0 + BK, na);  // (LOADcnt) while fragments load and WMMAs run
      loadB(k0 + BK, nb);
    }

    // ---- fragments per ISA 7.12.2 (wave32 layouts), 2x ds_load_b128 each
    FragU af[2], bf[4];
#pragma unroll
    for (int i = 0; i < 2; ++i) {
      int rowb = (wm + 16 * i + l15) * (LDSS / 2);   // u32 index of row start
      // A 16x32 f16: vgprs 0-3: K = {0..7}+8*kh ; vgprs 4-7: K = {16..23}+8*kh
      af[i].q[0] = *(const v4u*)&lA32[rowb + 4 * kh];
      af[i].q[1] = *(const v4u*)&lA32[rowb + 8 + 4 * kh];
    }
#pragma unroll
    for (int j = 0; j < 4; ++j) {
      // B 32x16 f16: lane -> column, vgprs 0-3: K = {0..7}+8*kh ; vgprs 4-7: K = {16..23}+8*kh
      int colb = (wn + 16 * j + l15) * (LDSS / 2) + 4 * kh;
      bf[j].q[0] = *(const v4u*)&lB32[colb];
      bf[j].q[1] = *(const v4u*)&lB32[colb + 8];
    }

    // ---- 8 WMMAs per wave per K-step
#pragma unroll
    for (int i = 0; i < 2; ++i)
#pragma unroll
      for (int j = 0; j < 4; ++j)
        acc[i][j] = __builtin_amdgcn_wmma_f32_16x16x32_f16(
            false, af[i].v, false, bf[j].v, (short)0, acc[i][j], false, false);

    __syncthreads();

    if (more) {
#pragma unroll
      for (int it = 0; it < 4; ++it) { ra[it] = na[it]; rb[it] = nb[it]; }
    }
  }

  // ---- epilogue: C/D layout: vgpr r -> row r + 8*kh, lane&15 -> col
#pragma unroll
  for (int i = 0; i < 2; ++i) {
#pragma unroll
    for (int j = 0; j < 4; ++j) {
      int colg = n0 + wn + 16 * j + l15;
      if (colg < N) {
        int rowb = m0 + wm + 16 * i + 8 * kh;
        float bv = bias ? bias[colg] : 0.0f;
#pragma unroll
        for (int r = 0; r < 8; ++r) {
          float v = acc[i][j][r] * alpha + bv;
          if (ACT == 1) { float rr = fmaxf(v, 0.0f); v = rr * rr; }  // relu^2
          C[(long long)(rowb + r) * ldc + colg] = v;
        }
      }
    }
  }
}

// ---------------------------------------------------------------------------
// Elementwise / small kernels
// ---------------------------------------------------------------------------
__global__ void k_embed(const float* __restrict__ emb, const int* __restrict__ x,
                        float* __restrict__ Hb)
{
  long long idx = (long long)blockIdx.x * 256 + threadIdx.x;
  if (idx >= (long long)B_ * S4_ * D_) return;
  int d  = (int)(idx % D_);
  long long t = idx / D_;
  int tt = (int)(t % S4_);
  int b  = (int)(t / S4_);
  int s = tt >> 2, ci = tt & 3;
  int byte = x[((long long)b * S_ + s) * 4 + ci];
  int j = d >> 1;
  float ang = (float)s * expf(-logf(10000.0f) * (float)(2 * j) / (float)D_);
  float pe = (d & 1) ? cosf(ang) : sinf(ang);
  Hb[idx] = emb[((long long)ci * 256 + byte) * D_ + d] + pe;
}

__global__ void k_dyt_attn(float* __restrict__ A, const float* __restrict__ a,
                           const float* __restrict__ w, const float* __restrict__ bb,
                           const float* __restrict__ depth)
{
  long long idx = (long long)blockIdx.x * 256 + threadIdx.x;
  if (idx >= (long long)B_ * S4_ * D_) return;
  int hd = (int)(idx % D_) & (HD_ - 1);
  float v = A[idx];
  A[idx] = (tanhf(a[0] * v) * w[hd] + bb[hd]) * depth[0];
}

__global__ void k_add_dyt(float* __restrict__ X, const float* __restrict__ R,
                          const float* __restrict__ a, const float* __restrict__ w,
                          const float* __restrict__ bb)
{
  long long idx = (long long)blockIdx.x * 256 + threadIdx.x;
  if (idx >= (long long)B_ * S4_ * D_) return;
  int d = (int)(idx % D_);
  float v = X[idx] + R[idx];
  X[idx] = tanhf(a[0] * v) * w[d] + bb[d];
}

// masked (tril) input -> depthwise 3x3 conv over (q,k), pad (2,0)/(1,1), +bias
__global__ void k_kqconv(const float* __restrict__ logits,  // (H,S4,S4) per batch
                         const float* __restrict__ w,       // (H,3,3)
                         const float* __restrict__ bb,      // (H)
                         float* __restrict__ out)           // (H,S4,S4)
{
  long long idx = (long long)blockIdx.x * 256 + threadIdx.x;
  if (idx >= (long long)H_ * S4_ * S4_) return;
  int kk = (int)(idx % S4_);
  int q  = (int)((idx / S4_) % S4_);
  int h  = (int)(idx / ((long long)S4_ * S4_));
  const float* L  = logits + (long long)h * S4_ * S4_;
  const float* wh = w + h * 9;
  float acc = bb[h];
#pragma unroll
  for (int i = 0; i < 3; ++i) {
    int qq = q - 2 + i;
    if (qq < 0) continue;
#pragma unroll
    for (int j = 0; j < 3; ++j) {
      int kj = kk - 1 + j;
      if (kj < 0 || kj >= S4_ || kj > qq) continue;   // tril mask applied pre-conv
      acc += wh[i * 3 + j] * L[(long long)qq * S4_ + kj];
    }
  }
  out[idx] = acc;
}

__global__ __launch_bounds__(256) void k_softmax(float* __restrict__ buf)
{
  float* row = buf + (long long)blockIdx.x * S4_;
  __shared__ float red[256];
  int tid = threadIdx.x;
  float m = -1e30f;
  for (int c = tid; c < S4_; c += 256) m = fmaxf(m, row[c]);
  red[tid] = m; __syncthreads();
  for (int s = 128; s > 0; s >>= 1) { if (tid < s) red[tid] = fmaxf(red[tid], red[tid + s]); __syncthreads(); }
  m = red[0]; __syncthreads();
  float sum = 0.f;
  for (int c = tid; c < S4_; c += 256) { float e = expf(row[c] - m); row[c] = e; sum += e; }
  red[tid] = sum; __syncthreads();
  for (int s = 128; s > 0; s >>= 1) { if (tid < s) red[tid] += red[tid + s]; __syncthreads(); }
  float inv = 1.0f / red[0];
  for (int c = tid; c < S4_; c += 256) row[c] *= inv;
}

// grouped head mix (k=1): mixed[h] = sum_i hm_w[h][i]*aw[group(h)*HG+i] + hm_b[h]
__global__ void k_headmix(const float* __restrict__ aw, const float* __restrict__ hw,
                          const float* __restrict__ hb, float* __restrict__ out)
{
  long long idx = (long long)blockIdx.x * 256 + threadIdx.x;
  if (idx >= (long long)H_ * S4_ * S4_) return;
  int h = (int)(idx / ((long long)S4_ * S4_));
  long long r = idx % ((long long)S4_ * S4_);
  int g = h / HG_;
  float acc = hb[h];
#pragma unroll
  for (int i2 = 0; i2 < HG_; ++i2)
    acc += hw[h * HG_ + i2] * aw[(long long)(g * HG_ + i2) * S4_ * S4_ + r];
  out[idx] = acc;
}

// conv1 (8,4,9,9) pad 4 over (B,4,S,D) view of H, then BN1
__global__ void k_conv1_bn(const float* __restrict__ Hb, const float* __restrict__ w,
                           const float* __restrict__ cb, const float* __restrict__ g,
                           const float* __restrict__ bb, const float* __restrict__ mm,
                           const float* __restrict__ vv, float* __restrict__ out)
{
  long long idx = (long long)blockIdx.x * 256 + threadIdx.x;
  if (idx >= (long long)B_ * 8 * S_ * D_) return;
  int d  = (int)(idx % D_);
  int s  = (int)((idx / D_) % S_);
  int oc = (int)((idx / ((long long)D_ * S_)) % 8);
  int b  = (int)(idx / ((long long)D_ * S_ * 8));
  const float* base = Hb + (long long)b * S4_ * D_;   // in[ic][s'][d'] = H[b][(s'*4+ic)][d']
  float acc = cb[oc];
  for (int ic = 0; ic < 4; ++ic)
    for (int di = 0; di < 9; ++di) {
      int ss = s - 4 + di;
      if (ss < 0 || ss >= S_) continue;
      const float* rowp = base + (long long)(ss * 4 + ic) * D_;
      for (int dj = 0; dj < 9; ++dj) {
        int dd = d - 4 + dj;
        if (dd < 0 || dd >= D_) continue;
        acc += w[((oc * 4 + ic) * 9 + di) * 9 + dj] * rowp[dd];
      }
    }
  float inv = rsqrtf(vv[oc] + 1e-5f);
  out[idx] = (acc - mm[oc]) * (g[oc] * inv) + bb[oc];
}

// conv2 (2,4,5,5) pad 2 over (B,4,S,2D) reinterpretation of conv1 out, BN2, relu^2
__global__ void k_conv2_bn_srelu(const float* __restrict__ Y, const float* __restrict__ w,
                                 const float* __restrict__ cb, const float* __restrict__ g,
                                 const float* __restrict__ bb, const float* __restrict__ mm,
                                 const float* __restrict__ vv, float* __restrict__ out)
{
  const int E = 2 * D_;  // 1024
  long long idx = (long long)blockIdx.x * 256 + threadIdx.x;
  if (idx >= (long long)B_ * 2 * S_ * E) return;
  int e  = (int)(idx % E);
  int s  = (int)((idx / E) % S_);
  int oc = (int)((idx / ((long long)E * S_)) % 2);
  int b  = (int)(idx / ((long long)E * S_ * 2));
  const float* base = Y + (long long)b * 8 * S_ * D_;  // (4,S,2D) flat view
  float acc = cb[oc];
  for (int ic = 0; ic < 4; ++ic)
    for (int di = 0; di < 5; ++di) {
      int ss = s - 2 + di;
      if (ss < 0 || ss >= S_) continue;
      const float* rowp = base + (long long)ic * S_ * E + (long long)ss * E;
      for (int dj = 0; dj < 5; ++dj) {
        int ee = e - 2 + dj;
        if (ee < 0 || ee >= E) continue;
        acc += w[((oc * 4 + ic) * 5 + di) * 5 + dj] * rowp[ee];
      }
    }
  float inv = rsqrtf(vv[oc] + 1e-5f);
  float v = (acc - mm[oc]) * (g[oc] * inv) + bb[oc];
  float rr = fmaxf(v, 0.0f);
  out[idx] = rr * rr;
}

__global__ void k_meanpool(const float* __restrict__ Z, float* __restrict__ P)
{
  const int E = 2 * D_;
  long long idx = (long long)blockIdx.x * 256 + threadIdx.x;
  if (idx >= (long long)B_ * S_ * E) return;
  int e = (int)(idx % E);
  int s = (int)((idx / E) % S_);
  int b = (int)(idx / ((long long)E * S_));
  long long base = (long long)b * 2 * S_ * E + (long long)s * E + e;
  P[idx] = 0.5f * (Z[base] + Z[base + (long long)S_ * E]);
}

// ---------------------------------------------------------------------------
// Host-side orchestration (stream-only; all scratch from d_ws).
// Workspace layout (floats): H 2M | Q 2M | K 2M | V 2M | AT 2M | FF 8M | LOG 8M | AW 8M
// Total 34M floats = 136 MB.
// ---------------------------------------------------------------------------
extern "C" void kernel_launch(void* const* d_in, const int* in_sizes, int n_in,
                              void* d_out, int out_size, void* d_ws, size_t ws_size,
                              hipStream_t stream)
{
  (void)in_sizes; (void)n_in; (void)out_size; (void)ws_size;
  auto f = [&](int i) { return (const float*)d_in[i]; };
  const float* emb = f(0);
  const int*   x   = (const int*)d_in[67];

  const long long M1 = 1024LL * 1024LL;
  float* ws   = (float*)d_ws;
  float* Hbuf = ws;
  float* QB   = ws + 2 * M1;
  float* KB   = ws + 4 * M1;
  float* VB   = ws + 6 * M1;
  float* AT   = ws + 8 * M1;
  float* FF   = ws + 10 * M1;
  float* LOG  = ws + 18 * M1;
  float* AW   = ws + 26 * M1;
  float* OUT1 = LOG;            // reuse after layers: 4.19M floats
  float* Z    = AW;             // 2.1M floats
  float* POOL = AW + 4 * M1;    // 1.05M floats

  const int NT = B_ * S4_;      // 4096 tokens
  dim3 blk(256);

  { long long n = (long long)NT * D_;
    k_embed<<<dim3((unsigned)((n + 255) / 256)), blk, 0, stream>>>(emb, x, Hbuf); }

  for (int l = 0; l < 2; ++l) {
    int p = 1 + l * 26;
    const float *qW = f(p + 0),  *qb = f(p + 1),  *kW = f(p + 2),  *kb = f(p + 3);
    const float *vW = f(p + 4),  *vb = f(p + 5),  *oW = f(p + 6),  *ob = f(p + 7);
    const float *kqw = f(p + 8), *kqb = f(p + 9), *hmw = f(p + 10), *hmb = f(p + 11);
    const float *dyta = f(p + 12), *dytw = f(p + 13), *dytb = f(p + 14), *depth = f(p + 15);
    const float *fw1 = f(p + 16), *fb1 = f(p + 17), *fw2 = f(p + 18), *fb2 = f(p + 19);
    const float *n1a = f(p + 20), *n1w = f(p + 21), *n1b = f(p + 22);
    const float *n2a = f(p + 23), *n2w = f(p + 24), *n2b = f(p + 25);

    dim3 gP(D_ / BN, NT / BM, 1);  // (4,32): 4096x512x512 projections
    k_gemm<0,false><<<gP, blk, 0, stream>>>(Hbuf, D_, 0, qW, D_, 0, QB, D_, 0, NT, D_, D_, qb, 1.f);
    k_gemm<0,false><<<gP, blk, 0, stream>>>(Hbuf, D_, 0, kW, D_, 0, KB, D_, 0, NT, D_, D_, kb, 1.f);
    k_gemm<0,false><<<gP, blk, 0, stream>>>(Hbuf, D_, 0, vW, D_, 0, VB, D_, 0, NT, D_, D_, vb, 1.f);

    for (int b = 0; b < B_; ++b) {  // per-b: score scratch (67MB) stays L2-resident
      const float* Qb = QB + (long long)b * S4_ * D_;
      const float* Kb = KB + (long long)b * S4_ * D_;
      const float* Vb = VB + (long long)b * S4_ * D_;
      // logits[h] = (1/8) * q @ k^T  (B-source is N x K row-major -> BNXK=true)
      dim3 gqk(S4_ / BN, S4_ / BM, H_);  // (8,8,8)
      k_gemm<0,true><<<gqk, blk, 0, stream>>>(Qb, D_, HD_, Kb, D_, HD_,
          LOG, S4_, (long long)S4_ * S4_, S4_, S4_, HD_, nullptr, 0.125f);
      long long ne = (long long)H_ * S4_ * S4_;
      k_kqconv<<<dim3((unsigned)((ne + 255) / 256)), blk, 0, stream>>>(LOG, kqw, kqb, AW);
      k_softmax<<<dim3(H_ * S4_), blk, 0, stream>>>(AW);
      k_headmix<<<dim3((unsigned)((ne + 255) / 256)), blk, 0, stream>>>(AW, hmw, hmb, LOG);
      // out[h] = mixed @ v, scattered into (t, h*HD+hd) layout via ldc/strides
      dim3 gav(1, S4_ / BM, H_);  // N=64 handled by guards
      k_gemm<0,false><<<gav, blk, 0, stream>>>(LOG, S4_, (long long)S4_ * S4_,
          Vb, D_, HD_, AT + (long long)b * S4_ * D_, D_, HD_, S4_, HD_, S4_, nullptr, 1.f);
    }

    { long long n = (long long)NT * D_;
      k_dyt_attn<<<dim3((unsigned)((n + 255) / 256)), blk, 0, stream>>>(AT, dyta, dytw, dytb, depth); }
    k_gemm<0,false><<<gP, blk, 0, stream>>>(AT, D_, 0, oW, D_, 0, QB, D_, 0, NT, D_, D_, ob, 1.f);
    { long long n = (long long)NT * D_;
      k_add_dyt<<<dim3((unsigned)((n + 255) / 256)), blk, 0, stream>>>(Hbuf, QB, n1a, n1w, n1b); }

    dim3 gF1(DFF_ / BN, NT / BM, 1);
    k_gemm<1,false><<<gF1, blk, 0, stream>>>(Hbuf, D_, 0, fw1, DFF_, 0, FF, DFF_, 0, NT, DFF_, D_, fb1, 1.f);
    k_gemm<0,false><<<gP,  blk, 0, stream>>>(FF, DFF_, 0, fw2, D_, 0, QB, D_, 0, NT, D_, DFF_, fb2, 1.f);
    { long long n = (long long)NT * D_;
      k_add_dyt<<<dim3((unsigned)((n + 255) / 256)), blk, 0, stream>>>(Hbuf, QB, n2a, n2w, n2b); }
  }

  // final head
  const float *c1w = f(53), *c1b = f(54), *g1 = f(55), *b1 = f(56), *m1 = f(57), *v1 = f(58);
  const float *c2w = f(59), *c2b = f(60), *g2 = f(61), *b2 = f(62), *m2 = f(63), *v2 = f(64);
  const float *fcw = f(65), *fcb = f(66);

  { long long n = (long long)B_ * 8 * S_ * D_;
    k_conv1_bn<<<dim3((unsigned)((n + 255) / 256)), blk, 0, stream>>>(Hbuf, c1w, c1b, g1, b1, m1, v1, OUT1); }
  { long long n = (long long)B_ * 2 * S_ * (2 * D_);
    k_conv2_bn_srelu<<<dim3((unsigned)((n + 255) / 256)), blk, 0, stream>>>(OUT1, c2w, c2b, g2, b2, m2, v2, Z); }
  { long long n = (long long)B_ * S_ * (2 * D_);
    k_meanpool<<<dim3((unsigned)((n + 255) / 256)), blk, 0, stream>>>(Z, POOL); }

  dim3 gFC(EMB_ / BN, (B_ * S_) / BM, 1);
  k_gemm<0,false><<<gFC, blk, 0, stream>>>(POOL, 2 * D_, 0, fcw, EMB_, 0,
      (float*)d_out, EMB_, 0, B_ * S_, EMB_, 2 * D_, fcb, 1.f);
}
